// match_pyramid_29703993819504
// MI455X (gfx1250) — matrix operand order repaired and checked by reading, not compiled, not run on hardware
//
#include <hip/hip_runtime.h>
#include <stdint.h>

// MatchPyramid ranking net, fused for gfx1250 (MI455X, wave32, WMMA, async-LDS).
// K0: f32 -> f16 conversion of query + docs (pos first, then negs).
// K1: per (doc, 64-col stripe):
//       - doc stripe (80 cols x 64 dims f16, 10 KB) staged into LDS via
//         global_load_async_to_lds_b128 (ASYNCcnt), loaded ONCE per workgroup
//         instead of once per wave (8x traffic cut),
//       - m = Q·D^T via v_wmma_f32_16x16x32_f16 (A from global, B from LDS),
//       - conv1(3x3)+relu on the fly + streaming order-preserving top-8 over
//         the 126 rows -> pool1[B][8][8][512pad] in workspace.
// K2: per doc: top-8 over 510 cols, conv2(2x2)+relu, top-4/top-4, FC1+relu,
//     FC2+relu -> out[b].
//
// Workspace layout (needs ~25.2 MB):
//   [0,               16 KB)   query f16   (128x64)
//   [16 KB,     16K + 8 MB)    docs  f16   (128x512x64)
//   [+,             +16 MB)    pool1 f32   (128x8x8x512, cols 510/511 unused)

typedef __attribute__((ext_vector_type(16))) _Float16 v16h;
typedef __attribute__((ext_vector_type(8)))  _Float16 v8h;
typedef __attribute__((ext_vector_type(8)))  float    v8f;

#define NEG_INF (-3.402823466e38f)

union HFrag { v16h v; v8h h[2]; };

// ---- branchless streaming top-K (keeps earliest index on ties, like lax.top_k)
template<int K>
__device__ __forceinline__ void topk_update(float (&tv)[K], int (&ti)[K], float v, int r) {
  float mv = tv[0]; int ms = 0;
#pragma unroll
  for (int k = 1; k < K; ++k) { bool lt = tv[k] < mv; mv = lt ? tv[k] : mv; ms = lt ? k : ms; }
  bool take = v > mv;
#pragma unroll
  for (int k = 0; k < K; ++k) {
    bool rep = take && (k == ms);
    tv[k] = rep ? v : tv[k];
    ti[k] = rep ? r : ti[k];
  }
}

// ---- restore original-index order (reference sorts top-k indices ascending)
template<int K>
__device__ __forceinline__ void sort_by_index(float (&tv)[K], int (&ti)[K]) {
#pragma unroll
  for (int p = 0; p < K; ++p)
#pragma unroll
    for (int q = 0; q + 1 < K - p; ++q) {
      bool sw = ti[q] > ti[q + 1];
      float va = sw ? tv[q + 1] : tv[q];
      float vb = sw ? tv[q]     : tv[q + 1];
      int   ia = sw ? ti[q + 1] : ti[q];
      int   ib = sw ? ti[q]     : ti[q + 1];
      tv[q] = va; tv[q + 1] = vb; ti[q] = ia; ti[q + 1] = ib;
    }
}

// ===================== K0: f32 -> f16 packing =====================
__global__ __launch_bounds__(256) void cvt_kernel(
    const float* __restrict__ q, const float* __restrict__ pos,
    const float* __restrict__ neg, _Float16* __restrict__ qh,
    _Float16* __restrict__ dh) {
  const long NQ = 128L * 64;
  const long ND = 128L * 512 * 64;
  long stride = (long)gridDim.x * blockDim.x;
  for (long i = (long)blockIdx.x * blockDim.x + threadIdx.x; i < NQ + ND; i += stride) {
    if (i < NQ) {
      qh[i] = (_Float16)q[i];
    } else {
      long j = i - NQ;                       // ((b*512)+l)*64+d
      float v = (j < 32768L) ? pos[j] : neg[j - 32768L];  // b==0 -> pos doc
      dh[j] = (_Float16)v;
    }
  }
}

// ===================== K1: async-LDS stage + WMMA GEMM + conv1 + row top-8 ====
__global__ __launch_bounds__(256) void mm_conv_pool_kernel(
    const _Float16* __restrict__ qh, const _Float16* __restrict__ dh,
    const float* __restrict__ c1w, const float* __restrict__ c1b,
    float* __restrict__ pool1) {
  __shared__ float    mL[128][80];     // m stripe: 128 q-rows x 80 doc cols (40 KB)
  __shared__ _Float16 dstripe[80][64]; // doc stripe staged once per WG (10 KB)
  __shared__ float    w1s[80];         // conv1 weights (72) + bias (8)

  const int s   = blockIdx.x;     // stripe 0..7 -> m cols [64s, 64s+80)
  const int b   = blockIdx.y;     // doc
  const int c0  = s * 64;
  const int tid = threadIdx.x;

  // ---- async-stage doc stripe into LDS: 80 rows x 128 B = 640 x b128 chunks
  for (int i = tid; i < 640; i += 256) {
    const int col  = i >> 3;          // local stripe column 0..79
    const int part = i & 7;           // 16-byte chunk within the 128-B row
    int gcol = c0 + col; gcol = gcol > 511 ? 511 : gcol;  // pad cols: staged, never used
    const _Float16* gp = dh + ((long)b * 512 + gcol) * 64 + part * 8;
    unsigned lds = (unsigned)(uintptr_t)(&dstripe[col][part * 8]);
    asm volatile("global_load_async_to_lds_b128 %0, %1, off"
                 :: "v"(lds), "v"(gp) : "memory");
  }

  if (tid < 72)      w1s[tid] = c1w[tid];
  else if (tid < 80) w1s[tid] = c1b[tid - 72];

  // ---- stage A: m[rows 16w..16w+15][c0..c0+79] = Q * D^T via WMMA f16 ----
  const int  w    = tid >> 5;
  const int  lane = tid & 31;
  const int  lrow = lane & 15;
  const bool hi   = lane >= 16;

  // A fragment, 16x32 f16 per ISA layout: lane<16 -> K{0..7,16..23}; lane>=16 -> K{8..15,24..31}
  const int rowA = w * 16 + lrow;
  const _Float16* qp = qh + rowA * 64 + (hi ? 8 : 0);
  HFrag a0, a1;
  a0.h[0] = *(const v8h*)(qp);        // K chunk 0
  a0.h[1] = *(const v8h*)(qp + 16);
  a1.h[0] = *(const v8h*)(qp + 32);   // K chunk 1 (kc=32)
  a1.h[1] = *(const v8h*)(qp + 48);

  // wait for the async engine, then make the stripe visible to all waves
  asm volatile("s_wait_asynccnt 0x0" ::: "memory");
  __syncthreads();

#pragma unroll
  for (int t = 0; t < 5; ++t) {
    const int lcol = t * 16 + lrow;
    // B fragment, 32x16 f16: lane<16 -> K{0..7,16..23}; lane>=16 -> K{8..15,24..31} (matches A)
    const _Float16* dp = &dstripe[lcol][hi ? 8 : 0];
    HFrag b0, b1;
    b0.h[0] = *(const v8h*)(dp);
    b0.h[1] = *(const v8h*)(dp + 16);
    b1.h[0] = *(const v8h*)(dp + 32);
    b1.h[1] = *(const v8h*)(dp + 48);

    v8f c = {};
    c = __builtin_amdgcn_wmma_f32_16x16x32_f16(false, a0.v, false, b0.v, (short)0, c, false, false);
    c = __builtin_amdgcn_wmma_f32_16x16x32_f16(false, a1.v, false, b1.v, (short)0, c, false, false);

    const int rbase = w * 16 + (hi ? 8 : 0);  // C/D: vgpr v -> M=v (lanes<16) / v+8 (lanes>=16)
#pragma unroll
    for (int v = 0; v < 8; ++v) mL[rbase + v][t * 16 + lrow] = c[v];
  }
  __syncthreads();

  // ---- stage B: conv1 3x3 + relu, streaming top-8 over 126 rows per (ch,col) ----
  const int ncols  = (s == 7) ? 62 : 64;     // output cols are 0..509
  const int npairs = 8 * ncols;
  for (int p = tid; p < npairs; p += 256) {
    const int ch = p & 7;
    const int j  = p >> 3;                   // local output col; global col = c0 + j
    float wv[9];
#pragma unroll
    for (int i = 0; i < 9; ++i) wv[i] = w1s[ch * 9 + i];
    const float bias = w1s[72 + ch];

    float tv[8]; int ti[8];
#pragma unroll
    for (int k = 0; k < 8; ++k) { tv[k] = NEG_INF; ti[k] = 0x7fffffff; }

    for (int r = 0; r < 126; ++r) {
      float acc = bias;
#pragma unroll
      for (int u = 0; u < 3; ++u)
#pragma unroll
        for (int vv = 0; vv < 3; ++vv)
          acc += wv[u * 3 + vv] * mL[r + u][j + vv];
      acc = fmaxf(acc, 0.0f);
      topk_update<8>(tv, ti, acc, r);
    }
    sort_by_index<8>(tv, ti);

    float* dst = pool1 + (((long)b * 8 + ch) * 8) * 512 + (c0 + j);
#pragma unroll
    for (int k = 0; k < 8; ++k) dst[(long)k * 512] = tv[k];
  }
}

// ===================== K2: col top-8, conv2, top-4 pools, FC head =====================
__global__ __launch_bounds__(128) void head_kernel(
    const float* __restrict__ pool1,
    const float* __restrict__ c2w, const float* __restrict__ c2b,
    const float* __restrict__ l1w, const float* __restrict__ l1b,
    const float* __restrict__ l2w, const float* __restrict__ l2b,
    float* __restrict__ out) {
  const int b = blockIdx.x;
  const int t = threadIdx.x;

  __shared__ float h2[8][8][8];     // after 2-D k-max pool #1
  __shared__ float cc[16][7][7];    // conv2 + relu
  __shared__ float p3[16][4][7];    // pool rows -> 4
  __shared__ float feat[256];       // pool cols -> 4, flattened [oc][r][c]
  __shared__ float zbuf[100];
  __shared__ float w2s[528];        // conv2 w (512) + b (16)

  for (int i = t; i < 512; i += 128) w2s[i] = c2w[i];
  if (t < 16) w2s[512 + t] = c2b[t];

  // phase 1: top-8 over 510 columns, order preserved
  if (t < 64) {
    const int ch = t >> 3, kk = t & 7;
    const float* src = pool1 + (((long)b * 8 + ch) * 8 + kk) * 512;
    float tv[8]; int ti[8];
#pragma unroll
    for (int k = 0; k < 8; ++k) { tv[k] = NEG_INF; ti[k] = 0x7fffffff; }
    for (int c = 0; c < 510; ++c) topk_update<8>(tv, ti, src[c], c);
    sort_by_index<8>(tv, ti);
#pragma unroll
    for (int k = 0; k < 8; ++k) h2[ch][kk][k] = tv[k];
  }
  __syncthreads();

  // phase 2: conv2 2x2 (8 -> 16 ch) + relu -> [16][7][7]
  for (int i = t; i < 784; i += 128) {
    const int oc = i / 49, rem = i % 49, r = rem / 7, c = rem % 7;
    float acc = w2s[512 + oc];
#pragma unroll
    for (int ic = 0; ic < 8; ++ic)
#pragma unroll
      for (int u = 0; u < 2; ++u)
#pragma unroll
        for (int v = 0; v < 2; ++v)
          acc += w2s[((oc * 8 + ic) * 2 + u) * 2 + v] * h2[ic][r + u][c + v];
    cc[oc][r][c] = fmaxf(acc, 0.0f);
  }
  __syncthreads();

  // phase 3: top-4 over rows (7) per (oc, col)
  if (t < 112) {
    const int oc = t / 7, c = t % 7;
    float tv[4]; int ti[4];
#pragma unroll
    for (int k = 0; k < 4; ++k) { tv[k] = NEG_INF; ti[k] = 0x7fffffff; }
    for (int r = 0; r < 7; ++r) topk_update<4>(tv, ti, cc[oc][r][c], r);
    sort_by_index<4>(tv, ti);
#pragma unroll
    for (int k = 0; k < 4; ++k) p3[oc][k][c] = tv[k];
  }
  __syncthreads();

  // phase 4: top-4 over cols (7) per (oc, row) -> feat[oc*16 + r*4 + k]
  if (t < 64) {
    const int oc = t >> 2, r = t & 3;
    float tv[4]; int ti[4];
#pragma unroll
    for (int k = 0; k < 4; ++k) { tv[k] = NEG_INF; ti[k] = 0x7fffffff; }
    for (int c = 0; c < 7; ++c) topk_update<4>(tv, ti, p3[oc][r][c], c);
    sort_by_index<4>(tv, ti);
#pragma unroll
    for (int k = 0; k < 4; ++k) feat[oc * 16 + r * 4 + k] = tv[k];
  }
  __syncthreads();

  // phase 5: FC1 256 -> 100 + relu
  if (t < 100) {
    float acc = l1b[t];
    const float* wr = l1w + (long)t * 256;
    for (int i = 0; i < 256; ++i) acc += feat[i] * wr[i];
    zbuf[t] = fmaxf(acc, 0.0f);
  }
  __syncthreads();

  // phase 6: FC2 100 -> 1 + relu
  if (t == 0) {
    float acc = l2b[0];
    for (int j = 0; j < 100; ++j) acc += zbuf[j] * l2w[j];
    out[b] = fmaxf(acc, 0.0f);
  }
}

extern "C" void kernel_launch(void* const* d_in, const int* in_sizes, int n_in,
                              void* d_out, int out_size, void* d_ws, size_t ws_size,
                              hipStream_t stream) {
  (void)in_sizes; (void)n_in; (void)out_size; (void)ws_size;
  const float* query    = (const float*)d_in[0];
  const float* pos_doc  = (const float*)d_in[1];
  const float* neg_docs = (const float*)d_in[2];
  const float* c1w = (const float*)d_in[3];
  const float* c1b = (const float*)d_in[4];
  const float* c2w = (const float*)d_in[5];
  const float* c2b = (const float*)d_in[6];
  const float* l1w = (const float*)d_in[7];
  const float* l1b = (const float*)d_in[8];
  const float* l2w = (const float*)d_in[9];
  const float* l2b = (const float*)d_in[10];
  float* out = (float*)d_out;

  char* ws = (char*)d_ws;
  _Float16* qh    = (_Float16*)ws;                        // 16 KB
  _Float16* dh    = (_Float16*)(ws + 16384);              // 8 MB
  float*    pool1 = (float*)(ws + 16384 + 8388608);       // 16 MB

  cvt_kernel<<<2048, 256, 0, stream>>>(query, pos_doc, neg_docs, qh, dh);

  dim3 g1(8, 128);  // (stripe, doc)
  mm_conv_pool_kernel<<<g1, 256, 0, stream>>>(qh, dh, c1w, c1b, pool1);

  head_kernel<<<128, 128, 0, stream>>>(pool1, c2w, c2b, l1w, l1b, l2w, l2b, out);
}
